// LIIF_17076789968914
// MI455X (gfx1250) — matrix operand order repaired and checked by reading, not compiled, not run on hardware
//
#include <hip/hip_runtime.h>
#include <hip/hip_bf16.h>

// ---------------------------------------------------------------------------
// LIIF forward on gfx1250: bf16 WMMA MLP (580->256->256->256->256->3), fp32 acc
// MT=32 queries/block, 8 waves, each wave owns a 2x2 grid of 16x16 tiles.
// feat pre-transposed to channels-last bf16; layer-0 K-dim permuted to match.
// ---------------------------------------------------------------------------

typedef __attribute__((ext_vector_type(8)))  unsigned short us8;
typedef __attribute__((ext_vector_type(16))) unsigned short us16;
typedef __attribute__((ext_vector_type(16))) __bf16        v16bf;
typedef __attribute__((ext_vector_type(8)))  float         v8f;

#define BATCH 4
#define CH    64
#define HH    48
#define WW    48
#define IN_D  580
#define K0    608          // IN_D padded to multiple of 32
#define HID   256
#define MT    32           // queries per block

// workspace layout (ushort units), all bf16
#define OFF_W0T 0                   // [256 x 608], K permuted: k' = p*64 + c
#define OFF_W1T (256*608)           // [256 x 256]
#define OFF_W2T (OFF_W1T + 256*256)
#define OFF_W3T (OFF_W2T + 256*256)
#define OFF_W4T (OFF_W3T + 256*256) // [16 x 256] (cols>=3 zero)
#define WTOTAL  (OFF_W4T + 16*256)
#define OFF_FEATT WTOTAL            // [B][48][48][64] channels-last bf16
#define FEATT_LEN (BATCH*HH*WW*CH)

__device__ __forceinline__ unsigned short f2bf(float f) {
    unsigned int u = __float_as_uint(f);
    u += 0x7FFFu + ((u >> 16) & 1u);   // round-to-nearest-even
    return (unsigned short)(u >> 16);
}

__device__ __forceinline__ us8 ld8(const unsigned short* p) {
    return *(const us8*)p;
}

__device__ __forceinline__ v16bf mk16(us8 lo, us8 hi) {
    us16 t = __builtin_shufflevector(lo, hi, 0,1,2,3,4,5,6,7,8,9,10,11,12,13,14,15);
    return __builtin_bit_cast(v16bf, t);
}

// ---------------------------------------------------------------------------
// Pre-pass 1: fp32 weights -> bf16, transposed to [N x K], K zero-padded.
// Layer-0 K permuted to position-major: k' = p*64 + c  ->  orig k = c*9 + p.
// ---------------------------------------------------------------------------
__global__ void convert_weights(const float* __restrict__ W0,
                                const float* __restrict__ W1,
                                const float* __restrict__ W2,
                                const float* __restrict__ W3,
                                const float* __restrict__ W4,
                                unsigned short* __restrict__ ws) {
    for (int i = blockIdx.x * blockDim.x + threadIdx.x; i < WTOTAL;
         i += gridDim.x * blockDim.x) {
        float v = 0.0f;
        if (i < OFF_W1T) {                       // W0T [256 x 608], permuted K
            int n = i / K0, k = i % K0;
            if (k < 576) {
                int p = k >> 6, c = k & 63;      // permuted -> original
                v = W0[(c * 9 + p) * 256 + n];
            } else if (k < IN_D) {               // rel / rel_cell lanes unchanged
                v = W0[k * 256 + n];
            }
        } else if (i < OFF_W4T) {                // W1T/W2T/W3T [256 x 256]
            int i2 = i - OFF_W1T;
            int layer = i2 / (256 * 256);
            int r = i2 % (256 * 256);
            int n = r / 256, k = r % 256;
            const float* Wp = (layer == 0) ? W1 : (layer == 1) ? W2 : W3;
            v = Wp[k * 256 + n];
        } else {                                 // W4T [16 x 256]
            int i3 = i - OFF_W4T;
            int n = i3 / 256, k = i3 % 256;
            if (n < 3) v = W4[k * 3 + n];
        }
        ws[i] = f2bf(v);
    }
}

// ---------------------------------------------------------------------------
// Pre-pass 2: feat [B,C,H,W] fp32 -> featT [B,H,W,C] bf16 (channels-last)
// ---------------------------------------------------------------------------
__global__ void prep_feat(const float* __restrict__ feat,
                          unsigned short* __restrict__ ws) {
    for (int i = blockIdx.x * blockDim.x + threadIdx.x; i < FEATT_LEN;
         i += gridDim.x * blockDim.x) {
        int c = i & 63;
        int rest = i >> 6;                 // ((bb*H + y)*W + x)
        int x = rest % WW; rest /= WW;
        int y = rest % HH;
        int bb = rest / HH;
        ws[OFF_FEATT + i] = f2bf(feat[((bb * CH + c) * HH + y) * WW + x]);
    }
}

// ---------------------------------------------------------------------------
// One MLP layer: [32 x KD] @ [KD x 256] + bias, optional ReLU, bf16 out.
// Wave w owns N-tiles {2w, 2w+1} x M-tiles {0,1}: 4 accumulators.
// A from LDS, B from global.
// ---------------------------------------------------------------------------
template <int KD, bool RELU>
__device__ __forceinline__ void mlp_layer(const unsigned short* __restrict__ act,   // LDS, ld=KD
                                          const unsigned short* __restrict__ Wt,    // global [256 x KD]
                                          const float* __restrict__ bias,
                                          unsigned short* __restrict__ actout,      // LDS, ld=256
                                          int wid, int lane) {
    const int lrow  = lane & 15;
    const int aoff  = (lane & 16) ? 16 : 0;   // A: lanes>=16 hold K 16-31
    const int boff  = (lane & 16) ? 16 : 0;   // B: lanes>=16 hold K 16-31
    const unsigned short* arow0 = act + lrow * KD;          // M-tile 0
    const unsigned short* arow1 = act + (16 + lrow) * KD;   // M-tile 1
    const unsigned short* bp0   = Wt + (2 * wid * 16 + lrow) * KD;
    const unsigned short* bp1   = Wt + ((2 * wid + 1) * 16 + lrow) * KD;

    v8f acc00 = {0.f,0.f,0.f,0.f,0.f,0.f,0.f,0.f};  // (mt0, nt0)
    v8f acc01 = {0.f,0.f,0.f,0.f,0.f,0.f,0.f,0.f};  // (mt0, nt1)
    v8f acc10 = {0.f,0.f,0.f,0.f,0.f,0.f,0.f,0.f};  // (mt1, nt0)
    v8f acc11 = {0.f,0.f,0.f,0.f,0.f,0.f,0.f,0.f};  // (mt1, nt1)

#pragma unroll 2
    for (int ks = 0; ks < KD / 32; ++ks) {
        const int ka = ks * 32 + aoff;
        const int kb = ks * 32 + boff;
        v16bf a0 = mk16(ld8(arow0 + ka), ld8(arow0 + ka + 8));
        v16bf a1 = mk16(ld8(arow1 + ka), ld8(arow1 + ka + 8));
        v16bf b0 = mk16(ld8(bp0 + kb),   ld8(bp0 + kb + 8));
        v16bf b1 = mk16(ld8(bp1 + kb),   ld8(bp1 + kb + 8));
        acc00 = __builtin_amdgcn_wmma_f32_16x16x32_bf16(false, a0, false, b0,
                                                        (short)0, acc00, false, false);
        acc10 = __builtin_amdgcn_wmma_f32_16x16x32_bf16(false, a1, false, b0,
                                                        (short)0, acc10, false, false);
        acc01 = __builtin_amdgcn_wmma_f32_16x16x32_bf16(false, a0, false, b1,
                                                        (short)0, acc01, false, false);
        acc11 = __builtin_amdgcn_wmma_f32_16x16x32_bf16(false, a1, false, b1,
                                                        (short)0, acc11, false, false);
    }

    const int rowbase = (lane & 16) ? 8 : 0;
    const int col0 = 2 * wid * 16 + lrow;
    const int col1 = col0 + 16;
    const float bb0 = bias[col0];
    const float bb1 = bias[col1];
#pragma unroll
    for (int r = 0; r < 8; ++r) {
        float v00 = acc00[r] + bb0;
        float v01 = acc01[r] + bb1;
        float v10 = acc10[r] + bb0;
        float v11 = acc11[r] + bb1;
        if (RELU) {
            v00 = fmaxf(v00, 0.f); v01 = fmaxf(v01, 0.f);
            v10 = fmaxf(v10, 0.f); v11 = fmaxf(v11, 0.f);
        }
        actout[(rowbase + r) * 256 + col0]      = f2bf(v00);
        actout[(rowbase + r) * 256 + col1]      = f2bf(v01);
        actout[(16 + rowbase + r) * 256 + col0] = f2bf(v10);
        actout[(16 + rowbase + r) * 256 + col1] = f2bf(v11);
    }
}

// ---------------------------------------------------------------------------
// Main kernel: one block = 32 queries of one batch image, all 4 shifts.
// ---------------------------------------------------------------------------
__global__ __launch_bounds__(256) void liif_kernel(
    const float* __restrict__ coord, const float* __restrict__ cell,
    const float* __restrict__ b0, const float* __restrict__ b1,
    const float* __restrict__ b2, const float* __restrict__ b3,
    const float* __restrict__ b4,
    const unsigned short* __restrict__ wsw,
    float* __restrict__ out, int Q) {

    __shared__ unsigned short Abuf[MT * K0];   // layer-0 input / even hidden acts
    __shared__ unsigned short Hbuf[MT * HID];  // odd hidden acts
    __shared__ int   srow[MT], scol[MT];
    __shared__ float sc0[MT], sc1[MT];           // original coords
    __shared__ float srel0[MT], srel1[MT];       // rel for current shift
    __shared__ float src0[MT], src1[MT];         // rel_cell
    __shared__ float swgt[4 * MT];               // blend weight per shift
    __shared__ float soacc[MT * 3];              // output accumulator

    const int tid  = threadIdx.x;
    const int wid  = tid >> 5;
    const int lane = tid & 31;
    const int QT   = (Q + MT - 1) / MT;
    const int b    = blockIdx.x / QT;
    const int q0   = (blockIdx.x % QT) * MT;

    const float rxy = 1.0f / 48.0f;
    const unsigned short* featT = wsw + OFF_FEATT + b * (HH * WW * CH);

    if (tid < MT * 3) soacc[tid] = 0.0f;

    // Per-query setup: coords, rel_cell, and the 4 reversed-area blend weights
    if (tid < MT) {
        const int q = tid, gq = q0 + q;
        if (gq < Q) {
            const float c0 = coord[(b * Q + gq) * 2 + 0];
            const float c1 = coord[(b * Q + gq) * 2 + 1];
            sc0[q] = c0; sc1[q] = c1;
            src0[q] = cell[(b * Q + gq) * 2 + 0] * 48.0f;
            src1[q] = cell[(b * Q + gq) * 2 + 1] * 48.0f;
            float areas[4];
#pragma unroll
            for (int s = 0; s < 4; ++s) {
                const float vx = (s & 2) ? 1.0f : -1.0f;
                const float vy = (s & 1) ? 1.0f : -1.0f;
                float cs0 = fminf(fmaxf(c0 + vx * rxy + 1e-6f, -1.0f + 1e-6f), 1.0f - 1e-6f);
                float cs1 = fminf(fmaxf(c1 + vy * rxy + 1e-6f, -1.0f + 1e-6f), 1.0f - 1e-6f);
                int r = (int)rintf((cs0 + 1.0f) * 24.0f - 0.5f); r = min(max(r, 0), 47);
                int c = (int)rintf((cs1 + 1.0f) * 24.0f - 0.5f); c = min(max(c, 0), 47);
                const float seqr = -1.0f + rxy + 2.0f * rxy * (float)r;
                const float seqc = -1.0f + rxy + 2.0f * rxy * (float)c;
                const float r0 = (c0 - seqr) * 48.0f;
                const float r1 = (c1 - seqc) * 48.0f;
                areas[s] = fabsf(r0 * r1) + 1e-9f;
            }
            const float tot = areas[0] + areas[1] + areas[2] + areas[3];
#pragma unroll
            for (int s = 0; s < 4; ++s) swgt[s * MT + q] = areas[3 - s] / tot;
        } else {
            sc0[q] = sc1[q] = 0.0f; src0[q] = src1[q] = 0.0f;
#pragma unroll
            for (int s = 0; s < 4; ++s) swgt[s * MT + q] = 0.0f;
        }
    }

    for (int s = 0; s < 4; ++s) {
        __syncthreads();  // protect Abuf reuse across iterations

        // index + rel computation for this shift
        if (tid < MT) {
            const int q = tid;
            const float vx = (s & 2) ? 1.0f : -1.0f;
            const float vy = (s & 1) ? 1.0f : -1.0f;
            float cs0 = fminf(fmaxf(sc0[q] + vx * rxy + 1e-6f, -1.0f + 1e-6f), 1.0f - 1e-6f);
            float cs1 = fminf(fmaxf(sc1[q] + vy * rxy + 1e-6f, -1.0f + 1e-6f), 1.0f - 1e-6f);
            int r = (int)rintf((cs0 + 1.0f) * 24.0f - 0.5f); r = min(max(r, 0), 47);
            int c = (int)rintf((cs1 + 1.0f) * 24.0f - 0.5f); c = min(max(c, 0), 47);
            srow[q] = r; scol[q] = c;
            const float seqr = -1.0f + rxy + 2.0f * rxy * (float)r;
            const float seqc = -1.0f + rxy + 2.0f * rxy * (float)c;
            srel0[q] = (sc0[q] - seqr) * 48.0f;
            srel1[q] = (sc1[q] - seqc) * 48.0f;
        }
        __syncthreads();

        // gather: 9 channels-last taps per query, pure 16B vector copies.
        // A-tile column order is position-major (matches permuted W0T).
        for (int e = tid; e < MT * 9 * 4; e += 256) {
            const int q = e / 36;
            const int r = e % 36;
            const int p = r >> 2;            // tap 0..8
            const int chunk = r & 3;         // 16 bf16 per chunk
            const int y = srow[q] + p / 3 - 1;
            const int x = scol[q] + p % 3 - 1;
            us8 v0 = {0,0,0,0,0,0,0,0};
            us8 v1 = {0,0,0,0,0,0,0,0};
            if ((unsigned)y < 48u && (unsigned)x < 48u) {
                const unsigned short* srcp = featT + (y * WW + x) * CH + chunk * 16;
                v0 = ld8(srcp);
                v1 = ld8(srcp + 8);
            }
            unsigned short* dst = &Abuf[q * K0 + p * 64 + chunk * 16];
            *(us8*)dst       = v0;
            *(us8*)(dst + 8) = v1;
        }
        for (int e = tid; e < MT * 32; e += 256) {
            const int q = e / 32, c2 = e % 32;
            float v = 0.0f;
            if      (c2 == 0) v = srel0[q];
            else if (c2 == 1) v = srel1[q];
            else if (c2 == 2) v = src0[q];
            else if (c2 == 3) v = src1[q];
            Abuf[q * K0 + 576 + c2] = f2bf(v);
        }
        __syncthreads();

        // MLP via WMMA, activations ping-pong Abuf <-> Hbuf
        mlp_layer<K0,  true>(Abuf, wsw + OFF_W0T, b0, Hbuf, wid, lane);
        __syncthreads();
        mlp_layer<HID, true>(Hbuf, wsw + OFF_W1T, b1, Abuf, wid, lane);
        __syncthreads();
        mlp_layer<HID, true>(Abuf, wsw + OFF_W2T, b2, Hbuf, wid, lane);
        __syncthreads();
        mlp_layer<HID, true>(Hbuf, wsw + OFF_W3T, b3, Abuf, wid, lane);
        __syncthreads();

        // final 256 -> 3 layer on waves 0/1 (one M-tile each), weighted accumulate
        if (wid < 2) {
            const int lrow = lane & 15;
            const int aoff = (lane & 16) ? 16 : 0;
            const int boff = (lane & 16) ? 16 : 0;
            const unsigned short* arow = Abuf + (wid * 16 + lrow) * HID;
            const unsigned short* bp   = wsw + OFF_W4T + lrow * HID;
            v8f acc = {0.f,0.f,0.f,0.f,0.f,0.f,0.f,0.f};
#pragma unroll 2
            for (int ks = 0; ks < HID / 32; ++ks) {
                v16bf a  = mk16(ld8(arow + ks * 32 + aoff), ld8(arow + ks * 32 + aoff + 8));
                v16bf bv = mk16(ld8(bp   + ks * 32 + boff), ld8(bp   + ks * 32 + boff + 8));
                acc = __builtin_amdgcn_wmma_f32_16x16x32_bf16(false, a, false, bv,
                                                              (short)0, acc, false, false);
            }
            const int col = lrow;
            if (col < 3) {
                const float bb = b4[col];
                const int rowbase = (lane & 16) ? 8 : 0;
#pragma unroll
                for (int r = 0; r < 8; ++r) {
                    const int q = wid * 16 + rowbase + r;
                    soacc[q * 3 + col] += (acc[r] + bb) * swgt[s * MT + q];
                }
            }
        }
    }
    __syncthreads();

    if (tid < MT) {
        const int gq = q0 + tid;
        if (gq < Q) {
#pragma unroll
            for (int n = 0; n < 3; ++n)
                out[(b * Q + gq) * 3 + n] = soacc[tid * 3 + n];
        }
    }
}

// ---------------------------------------------------------------------------
extern "C" void kernel_launch(void* const* d_in, const int* in_sizes, int n_in,
                              void* d_out, int out_size, void* d_ws, size_t ws_size,
                              hipStream_t stream) {
    // setup_inputs order:
    // 0 img, 1 feat, 2 coord, 3 cell, 4 W0, 5 b0, 6 W1, 7 b1,
    // 8 W2, 9 b2, 10 W3, 11 b3, 12 W4, 13 b4
    const float* feat  = (const float*)d_in[1];
    const float* coord = (const float*)d_in[2];
    const float* cell  = (const float*)d_in[3];
    const float* W0 = (const float*)d_in[4];  const float* b0 = (const float*)d_in[5];
    const float* W1 = (const float*)d_in[6];  const float* b1 = (const float*)d_in[7];
    const float* W2 = (const float*)d_in[8];  const float* b2 = (const float*)d_in[9];
    const float* W3 = (const float*)d_in[10]; const float* b3 = (const float*)d_in[11];
    const float* W4 = (const float*)d_in[12]; const float* b4 = (const float*)d_in[13];

    const int Q = in_sizes[2] / (BATCH * 2);  // coord is (B, Q, 2)
    unsigned short* ws16 = (unsigned short*)d_ws;

    convert_weights<<<(WTOTAL + 255) / 256, 256, 0, stream>>>(W0, W1, W2, W3, W4, ws16);
    prep_feat<<<(FEATT_LEN + 255) / 256, 256, 0, stream>>>(feat, ws16);

    const int QT = (Q + MT - 1) / MT;
    liif_kernel<<<BATCH * QT, 256, 0, stream>>>(coord, cell,
                                                b0, b1, b2, b3, b4,
                                                ws16, (float*)d_out, Q);
}
